// HybridLNNModel_86371792322661
// MI455X (gfx1250) — matrix operand order repaired and checked by reading, not compiled, not run on hardware
//
#include <hip/hip_runtime.h>
#include <hip/hip_bf16.h>

// Problem dims (fixed by the reference)
#define BDIM   256
#define SDIM   128
#define PDIM   8
#define HDIM   256
#define OUTDIM 256
#define STATE  (PDIM*HDIM)

#define BT     16               // batch rows per block (one wave-row M=16)
#define NWAVE  8                // waves per block; wave w owns output cols [32w,32w+32)
#define XSTR   (HDIM + 16)      // bf16 LDS row stride (pad: 544B rows -> conflict spread)
#define XBUF   (BT * XSTR)      // one x-tilde buffer (elements)

typedef __attribute__((ext_vector_type(16))) __bf16 v16bf;
typedef __attribute__((ext_vector_type(8)))  float  v8f;

union Frag32B { v16bf v; uint4 q[2]; };  // 16 bf16 = 32 bytes = one WMMA 16-bit operand per lane

// ---------------------------------------------------------------------------
// Pre-shuffle weights into per-lane B-fragment order for v_wmma_f32_16x16x32_bf16.
// B-matrix (32x16, KxN) lane layout (ISA 7.12.2): lanes 0-15 hold col n=lane,
// K=0..7,16..23 (elem e<8: K=e, e>=8: K=e+8); lanes 16-31 hold col n=lane-16, K=8..15,24..31.
// Fragment linear index (per chain p): (((c*16 + g16)*32 + lane)*16 + e)
//   k = 32*c + (lane<16 ? 0 : 8) + (e&7) + (e&8 ? 16 : 0) ;  n = 16*g16 + (lane&15)
// Wt: B[k][n] = Ww[p][n][k]      (pre[b,o] = sum_i x[b,i]*Ww[o,i])
// Wo: B[k][n] = out_w[n][p*H+k]  (per-chain slice of output projection)
// ---------------------------------------------------------------------------
__global__ void shuffle_weights(const float* __restrict__ Ww,
                                const float* __restrict__ out_w,
                                __bf16* __restrict__ Wt,
                                __bf16* __restrict__ Wo) {
  int idx  = blockIdx.x * blockDim.x + threadIdx.x;     // P * 65536 total
  int e    = idx & 15;
  int lane = (idx >> 4) & 31;
  int g16  = (idx >> 9) & 15;
  int c    = (idx >> 13) & 7;
  int p    = idx >> 16;
  int k = 32*c + ((lane < 16) ? 0 : 8) + (e & 7) + ((e & 8) ? 16 : 0);
  int n = 16*g16 + (lane & 15);
  Wt[idx] = (__bf16)Ww[((size_t)p*HDIM + n)*HDIM + k];
  Wo[idx] = (__bf16)out_w[(size_t)n*STATE + p*HDIM + k];
}

// d_out[b, s, o] = out_b[o]; chain partial GEMMs atomically accumulate on top.
__global__ void init_out(float* __restrict__ y, const float* __restrict__ out_b) {
  int idx = blockIdx.x * blockDim.x + threadIdx.x;      // B*(S-1)*OUT
  y[idx] = out_b[idx & (OUTDIM - 1)];
}

// ---------------------------------------------------------------------------
// Persistent per-(chain, batch-tile) RK4 scan. 256 threads = 8 waves.
// Wave w: C tile = rows 0..15 x cols [32w, 32w+32) (two 16x16 accumulators).
// LDS holds DOUBLE-BUFFERED bf16 x-tilde [2][16][256] (A-matrix row-major):
// each eval reads one buffer and writes the other -> only ONE barrier per
// eval (after the store). B-fragments (Wt + Wo) are loop-invariant and kept
// register-resident across all 127 steps (CDNA5 VGPR-MSB extended addressing).
// ---------------------------------------------------------------------------
__launch_bounds__(256, 1)
__global__ void ltc_scan(const float* __restrict__ inp,    // [B,S,P]
                         const float* __restrict__ times,  // [S]
                         const float* __restrict__ Wb,     // [P,H]
                         const float* __restrict__ Uw,     // [P,H]
                         const float* __restrict__ Ub,     // [P,H]
                         const float* __restrict__ bpar,   // [P,H]
                         const float* __restrict__ tau,    // [P,H]
                         const __bf16* __restrict__ Wt,    // preshuffled recurrent weights
                         const __bf16* __restrict__ Wo,    // preshuffled output weights
                         float* __restrict__ y)            // [B,S-1,OUT] (pre-init out_b)
{
  __shared__ __bf16 xt_s[2 * XBUF];
  __shared__ float  u_s[2 * BT];   // double-buffered per-step inputs

  const int tid  = threadIdx.x;
  const int lane = tid & 31;
  const int w    = tid >> 5;
  const int p    = blockIdx.x >> 4;          // 16 batch tiles per chain
  const int b0   = (blockIdx.x & 15) * BT;

  const int hi     = lane >> 4;              // C layout: lanes>=16 cover M+8
  const int l16    = lane & 15;
  const int m_base = 8 * hi;
  const int n0     = 32*w + l16;             // subtile-0 column
  const int n1     = n0 + 16;                // subtile-1 column

  // per-column chain parameters (2 columns per lane)
  const int pc = p * HDIM;
  const float uw0 = Uw[pc+n0],  uw1 = Uw[pc+n1];
  const float rt0 = 1.0f / tau[pc+n0], rt1 = 1.0f / tau[pc+n1];
  const float ca0 = Wb[pc+n0] + Ub[pc+n0] + bpar[pc+n0];
  const float ca1 = Wb[pc+n1] + Ub[pc+n1] + bpar[pc+n1];

  const __bf16* wtp = Wt + (size_t)p * HDIM * HDIM;
  const __bf16* wop = Wo + (size_t)p * HDIM * OUTDIM;

  // register state: index i = r + 8*sub  <->  (m = r+m_base, n = sub?n1:n0)
  float xb[16], xt[16], k1[16], k2[16], k3[16], k4[16];
  #pragma unroll
  for (int i = 0; i < 16; ++i) { xb[i] = 0.0f; xt[i] = 0.0f; }

  // 16x(N=32)xK=256 GEMM from LDS buffer `rdo`: stage ALL 8 A-fragments
  // (sched_barrier pins the loads ahead), then stream 16 back-to-back WMMAs
  // (D->C accumulation has no hazard; dscnt waits become incremental).
  auto gemm = [&](const __bf16* bptr, int rdo, v8f& c0, v8f& c1) {
    Frag32B a[8];
    #pragma unroll
    for (int c8 = 0; c8 < 8; ++c8) {
      const int k0 = 32*c8 + 8*hi;
      a[c8].q[0] = *reinterpret_cast<const uint4*>(&xt_s[rdo + l16*XSTR + k0]);
      a[c8].q[1] = *reinterpret_cast<const uint4*>(&xt_s[rdo + l16*XSTR + k0 + 16]);
    }
    __builtin_amdgcn_sched_barrier(0);       // keep all A-loads ahead of WMMAs
    v8f z = {};
    c0 = z; c1 = z;
    #pragma unroll
    for (int c8 = 0; c8 < 8; ++c8) {
      // loop-invariant B fragments: contiguous 32B/lane, register-resident
      const uint4* bp0 = reinterpret_cast<const uint4*>(bptr + (((size_t)(c8*16 + 2*w  )*32 + lane) << 4));
      const uint4* bp1 = reinterpret_cast<const uint4*>(bptr + (((size_t)(c8*16 + 2*w+1)*32 + lane) << 4));
      Frag32B bf0; bf0.q[0] = bp0[0]; bf0.q[1] = bp0[1];
      Frag32B bf1; bf1.q[0] = bp1[0]; bf1.q[1] = bp1[1];
      c0 = __builtin_amdgcn_wmma_f32_16x16x32_bf16(false, a[c8].v, false, bf0.v, (short)0, c0, false, false);
      c1 = __builtin_amdgcn_wmma_f32_16x16x32_bf16(false, a[c8].v, false, bf1.v, (short)0, c1, false, false);
    }
  };

  // one dxdt evaluation from buffer `rdo` (no barrier: wave-local math only)
  auto feval = [&](float* kout, int rdo, const float* ub) {
    v8f c0, c1;
    gemm(wtp, rdo, c0, c1);
    #pragma unroll
    for (int r = 0; r < 8; ++r) {
      const float u = ub[r + m_base];
      float pre0 = c0[r] + ca0 + u * uw0;
      float xv0  = xt[r];
      kout[r]    = (-xv0 * rt0 + tanhf(pre0)) * sqrtf(xv0 + 1.0f);
      float pre1 = c1[r] + ca1 + u * uw1;
      float xv1  = xt[r + 8];
      kout[r+8]  = (-xv1 * rt1 + tanhf(pre1)) * sqrtf(xv1 + 1.0f);
    }
  };

  auto store_xt = [&](int wro) {             // publish bf16 x-tilde, then ONE barrier
    #pragma unroll
    for (int r = 0; r < 8; ++r) {
      xt_s[wro + (r + m_base)*XSTR + n0] = (__bf16)xt[r];
      xt_s[wro + (r + m_base)*XSTR + n1] = (__bf16)xt[r + 8];
    }
    __syncthreads();
  };

  if (tid < BT) u_s[tid] = inp[((size_t)(b0 + tid)*SDIM + 0)*PDIM + p];  // u(0) -> ubuf[0]
  store_xt(0);                               // x(0) = 0 -> buffer 0 (+ sync covers u(0))

  for (int s = 0; s < SDIM - 1; ++s) {
    const float dtv = times[s+1] - times[s];
    const float dt3 = dtv * (1.0f/3.0f);
    const float dt8 = dtv * 0.125f;
    const float* ub = u_s + (s & 1) * BT;
    // prefetch u(s+1) into the other u buffer; >=4 barriers separate it from
    // its readers (next step's elementwise), and all readers of its previous
    // contents finished before the e1 barrier of this step.
    if (tid < BT && (s + 1) < (SDIM - 1))
      u_s[((s + 1) & 1) * BT + tid] = inp[((size_t)(b0 + tid)*SDIM + (s + 1))*PDIM + p];

    // 3/8-rule RK4 (x-tilde ping-pongs: e1 rd0/wr1, e2 rd1/wr0, e3 rd0/wr1,
    // e4 rd1/wr0, out rd0 -> parity is static across steps)
    feval(k1, 0, ub);
    #pragma unroll
    for (int i = 0; i < 16; ++i) xt[i] = xb[i] + dt3 * k1[i];
    store_xt(XBUF);

    feval(k2, XBUF, ub);
    #pragma unroll
    for (int i = 0; i < 16; ++i) xt[i] = xb[i] + dtv * k2[i] - dt3 * k1[i];
    store_xt(0);

    feval(k3, 0, ub);
    #pragma unroll
    for (int i = 0; i < 16; ++i) xt[i] = xb[i] + dtv * (k1[i] - k2[i] + k3[i]);
    store_xt(XBUF);

    feval(k4, XBUF, ub);
    #pragma unroll
    for (int i = 0; i < 16; ++i) {
      xb[i] += dt8 * (k1[i] + 3.0f*(k2[i] + k3[i]) + k4[i]);
      xt[i] = xb[i];
    }
    store_xt(0);                             // publish x_new -> buffer 0

    // per-chain partial of output projection; accumulate via f32 atomics
    v8f y0, y1;
    gemm(wop, 0, y0, y1);
    #pragma unroll
    for (int r = 0; r < 8; ++r) {
      const int m = r + m_base;
      float* yrow = y + ((size_t)(b0 + m)*(SDIM - 1) + s)*OUTDIM;
      atomicAdd(yrow + n0, y0[r]);
      atomicAdd(yrow + n1, y1[r]);
    }
    // no barrier needed: e1's store targets buffer 1, and the e1 barrier of
    // the next iteration fences buffer-0 reuse
  }
}

extern "C" void kernel_launch(void* const* d_in, const int* in_sizes, int n_in,
                              void* d_out, int out_size, void* d_ws, size_t ws_size,
                              hipStream_t stream) {
  const float* inp   = (const float*)d_in[0];
  const float* times = (const float*)d_in[1];
  const float* Ww    = (const float*)d_in[2];
  const float* Wb    = (const float*)d_in[3];
  const float* Uw    = (const float*)d_in[4];
  const float* Ub    = (const float*)d_in[5];
  const float* bpar  = (const float*)d_in[6];
  const float* tau   = (const float*)d_in[7];
  const float* out_w = (const float*)d_in[8];
  const float* out_b = (const float*)d_in[9];
  float* y = (float*)d_out;

  __bf16* Wt = (__bf16*)d_ws;                                        // 1 MB
  __bf16* Wo = (__bf16*)((char*)d_ws + (size_t)PDIM*HDIM*HDIM*2);    // 1 MB

  shuffle_weights<<<(PDIM*HDIM*HDIM)/256, 256, 0, stream>>>(Ww, out_w, Wt, Wo);
  init_out<<<(BDIM*(SDIM-1)*OUTDIM)/256, 256, 0, stream>>>(y, out_b);
  ltc_scan<<<PDIM*(BDIM/BT), 256, 0, stream>>>(inp, times, Wb, Uw, Ub, bpar, tau, Wt, Wo, y);
}
